// MixedRON_10084583211210
// MI455X (gfx1250) — matrix operand order repaired and checked by reading, not compiled, not run on hardware
//
#include <hip/hip_runtime.h>
#include <hip/hip_bf16.h>
#include <math.h>

// ---------------------------------------------------------------------------
// MixedRON forward for MI455X (gfx1250, wave32, WMMA + TDM).
//
// Per step: f = tanh([x_t | hy] @ [x2h; h2h] + bias) -> elementwise updates.
// GEMM M=64, N=2048, K=2304 via v_wmma_f32_16x16x32_bf16 (f32 accumulate).
// Weights pre-packed (bf16) into the wave32 B-operand tile layout, grouped so
// each block's per-chunk panel (8 k-tiles x 4 n-tiles = 32 KB) is contiguous.
// The Tensor Data Mover streams panels into LDS (double-buffered, TENSORcnt),
// overlapped with WMMA compute; waves read B via ds_load.
// hy bf16 A-operand is double-buffered across the 512 step launches.
// Workspace: ~10.5 MB.
// ---------------------------------------------------------------------------

typedef __attribute__((ext_vector_type(16))) __bf16 v16bf;
typedef __attribute__((ext_vector_type(8)))  __bf16 v8bf;
typedef __attribute__((ext_vector_type(8)))  float  v8f;
typedef unsigned int v4u __attribute__((ext_vector_type(4)));
typedef int          v8i __attribute__((ext_vector_type(8)));
typedef int          v4i __attribute__((ext_vector_type(4)));

#define NIN   256
#define NH    2048
#define PP    1024
#define BB    64
#define TT    512
#define KTILES 72             // 2304 / 32
#define NBLOCKS 32            // NH / 64 columns per block
#define NCHUNK  9             // KTILES / 8
#define CHUNK_KT 8
#define CHUNK_DW 8192         // 8 kt * 4 nt * 512 bf16 = 32 KB = 8192 dwords

#define DT_C    0.042f
#define GAMMA_C 2.7f
#define EPS_C   4.7f

// ws layout (bytes)
#define WPACK_BYTES ((size_t)NBLOCKS * KTILES * 4 * 512 * 2)  // 9,437,184
#define HYB_BYTES   ((size_t)BB * NH * 2)                     // 262,144
#define HYF_BYTES   ((size_t)BB * NH * 4)                     // 524,288

#define OUT_SEC ((size_t)BB * TT * PP)                        // 33,554,432 floats

// ---------------------------------------------------------------------------
// Pack [x2h; h2h] (fp32 row-major KxN) into bf16 WMMA B tiles, grouped as
// wpack[blk(32)][kt(72)][nts(4)][512]:
//   element (lane, e) of tile = W[kt*32 + (lane>>4)*8 + (e&7) + (e>>3)*16][blk*64 + nts*16 + (lane&15)]
// (ISA 16-bit B-matrix 32x16 wave32 layout). Each block+chunk panel of
// 8 k-tiles is one contiguous 32 KB region -> single 1-D TDM copy.
// ---------------------------------------------------------------------------
__global__ __launch_bounds__(256) void pack_weights_kernel(
    const float* __restrict__ x2h, const float* __restrict__ h2h,
    __bf16* __restrict__ wpack)
{
  size_t total = (size_t)NBLOCKS * KTILES * 4 * 512;
  for (size_t idx = (size_t)blockIdx.x * blockDim.x + threadIdx.x;
       idx < total; idx += (size_t)gridDim.x * blockDim.x) {
    size_t tile = idx >> 9;
    int li   = (int)(idx & 511);
    int lane = li >> 4;
    int e    = li & 15;
    int nts  = (int)(tile & 3);
    int kt   = (int)((tile >> 2) % KTILES);
    int blk  = (int)(tile / (4 * KTILES));
    int k    = kt * 32 + (lane >> 4) * 8 + (e & 7) + (e >> 3) * 16;
    int n    = blk * 64 + nts * 16 + (lane & 15);
    float v  = (k < NIN) ? x2h[(size_t)k * NH + n]
                         : h2h[(size_t)(k - NIN) * NH + n];
    wpack[idx] = (__bf16)v;
  }
}

__global__ __launch_bounds__(256) void init_state_kernel(
    float* __restrict__ hyf, float* __restrict__ hzf, __bf16* __restrict__ hyb0)
{
  int idx = blockIdx.x * blockDim.x + threadIdx.x;   // 512*256 threads
  if (idx < BB * NH) {
    hyf[idx]  = 0.0f;
    hyb0[idx] = (__bf16)0.0f;
  }
  if (idx < BB * PP) hzf[idx] = 0.0f;
}

// ---------------------------------------------------------------------------
// TDM: issue a 1-D tensor_load_to_lds of one 32 KB weight panel.
// D# per CDNA5 ISA 8.3/8.4: group0 {count=1, lds_addr, global_addr, type=2},
// group1 {data_size=4B, tensor_dim0=tile_dim0=8192 dwords, stride=8192}.
// ---------------------------------------------------------------------------
__device__ __forceinline__ void tdm_issue_panel(const __bf16* wpack, int blk,
                                                int chunk, unsigned lds_byte_addr)
{
  unsigned long long gaddr = (unsigned long long)(uintptr_t)
      (wpack + ((size_t)(blk * KTILES + chunk * CHUNK_KT) * 4) * 512);

  v4u g0;
  g0.x = 1u;                                   // count=1 (valid), user mode
  g0.y = lds_byte_addr;                        // LDS destination (bytes)
  g0.z = (unsigned)gaddr;                      // global_addr[31:0]
  g0.w = (unsigned)((gaddr >> 32) & 0x1FFFFFFu) | (2u << 30);  // [56:32] | type=2

  v8i g1;
  g1[0] = (int)(2u << 16);                     // data_size = 4 bytes
  g1[1] = (int)((CHUNK_DW & 0xFFFFu) << 16);   // tensor_dim0[15:0]
  g1[2] = (int)((CHUNK_DW >> 16) | (1u << 16)); // tensor_dim0[31:16], tensor_dim1=1
  g1[3] = (int)((CHUNK_DW & 0xFFFFu) << 16);   // tile_dim0 = 8192
  g1[4] = 0;                                   // tile_dim1/2 unused
  g1[5] = (int)CHUNK_DW;                       // tensor_dim0_stride[31:0]
  g1[6] = 0;
  g1[7] = 0;

  v4i z4 = {0, 0, 0, 0};
#if __has_include(<hip/amd_detail/amd_gfx1250_TDM.h>)
  v8i z8 = {0, 0, 0, 0, 0, 0, 0, 0};
  __builtin_amdgcn_tensor_load_to_lds(g0, g1, z4, z4, z8, 0);
#else
  __builtin_amdgcn_tensor_load_to_lds(g0, g1, z4, z4, 0);
#endif
}

// ---------------------------------------------------------------------------
// One recurrent timestep.
// Grid: 32 blocks x 256 threads. Block b owns columns [b*64, b*64+64).
// Wave w: mt = w&3 (16 rows), nsub = w>>2 (two 16-col tiles) -> 16x32 output.
// B panels staged global->LDS by the TDM (double-buffered per 8-k-tile chunk).
// ---------------------------------------------------------------------------
__global__ __launch_bounds__(256) void step_kernel(
    int t,
    const float*  __restrict__ x,        // [B, T, NIN]
    const __bf16* __restrict__ wpack,    // packed B panels
    const float*  __restrict__ bias,     // [NH]
    const __bf16* __restrict__ hyb_in,   // [B, NH] bf16 (read)
    __bf16*       __restrict__ hyb_out,  // [B, NH] bf16 (write)
    float*        __restrict__ hyf,      // [B, NH] fp32 state
    float*        __restrict__ hzf,      // [B, PP] fp32 state
    float*        __restrict__ out)      // 4 x [B, T, PP]
{
  __shared__ __align__(32) __bf16 ldsB[2][CHUNK_KT][4][512];   // 64 KB

  const int lane = threadIdx.x & 31;
  const int wv   = threadIdx.x >> 5;
  const int mt   = wv & 3;
  const int nsub = wv >> 2;
  const int m0   = mt * 16;
  const int n0   = blockIdx.x * 64 + nsub * 32;  // two 16-wide tiles: n0, n0+16

  const int arow  = m0 + (lane & 15);            // A-operand row (batch index)
  const int khalf = (lane >> 4) * 8;             // A-operand K-half select

  const float*  xrow  = x + ((size_t)arow * TT + t) * NIN;
  const __bf16* hyrow = hyb_in + (size_t)arow * NH;

  v8f c0 = {};
  v8f c1 = {};

  const unsigned lds0 = (unsigned)(uintptr_t)&ldsB[0][0][0][0];
  const unsigned lds1 = (unsigned)(uintptr_t)&ldsB[1][0][0][0];

  // Prime the pipeline: chunk 0 (the x/x2h K-range) into buffer 0.
  if (wv == 0) tdm_issue_panel(wpack, blockIdx.x, 0, lds0);

  // ---- Chunk 0: A from x (fp32 -> bf16 on the fly) ----
  if (wv == 0) {
    tdm_issue_panel(wpack, blockIdx.x, 1, lds1);
    __builtin_amdgcn_s_wait_tensorcnt(1);        // chunk 0 resident
  }
  __syncthreads();
#pragma unroll
  for (int kl = 0; kl < CHUNK_KT; ++kl) {
    v16bf a;
    const float* xp = xrow + kl * 32 + khalf;
#pragma unroll
    for (int i = 0; i < 8; ++i) a[i]     = (__bf16)xp[i];
#pragma unroll
    for (int i = 0; i < 8; ++i) a[8 + i] = (__bf16)xp[16 + i];
    v16bf b0 = *(const v16bf*)&ldsB[0][kl][nsub * 2][lane * 16];
    v16bf b1 = *(const v16bf*)&ldsB[0][kl][nsub * 2 + 1][lane * 16];
    c0 = __builtin_amdgcn_wmma_f32_16x16x32_bf16(false, a, false, b0,
                                                 (short)0, c0, false, false);
    c1 = __builtin_amdgcn_wmma_f32_16x16x32_bf16(false, a, false, b1,
                                                 (short)0, c1, false, false);
  }
  __syncthreads();

  // ---- Chunks 1..8: A from recurrent hy (bf16), steady-state pipeline ----
  for (int c = 1; c < NCHUNK; ++c) {
    if (wv == 0) {
      if (c + 1 < NCHUNK) {
        tdm_issue_panel(wpack, blockIdx.x, c + 1, (c & 1) ? lds0 : lds1);
        __builtin_amdgcn_s_wait_tensorcnt(1);    // chunk c resident
      } else {
        __builtin_amdgcn_s_wait_tensorcnt(0);
      }
    }
    __syncthreads();
    const int cb = c & 1;
    const __bf16* hbase = hyrow + (size_t)(c - 1) * CHUNK_KT * 32 + khalf;
#pragma unroll
    for (int kl = 0; kl < CHUNK_KT; ++kl) {
      const __bf16* hp = hbase + kl * 32;
      v8bf lo = *(const v8bf*)(hp);
      v8bf hi = *(const v8bf*)(hp + 16);
      v16bf a = __builtin_shufflevector(lo, hi,
                    0,1,2,3,4,5,6,7,8,9,10,11,12,13,14,15);
      v16bf b0 = *(const v16bf*)&ldsB[cb][kl][nsub * 2][lane * 16];
      v16bf b1 = *(const v16bf*)&ldsB[cb][kl][nsub * 2 + 1][lane * 16];
      c0 = __builtin_amdgcn_wmma_f32_16x16x32_bf16(false, a, false, b0,
                                                   (short)0, c0, false, false);
      c1 = __builtin_amdgcn_wmma_f32_16x16x32_bf16(false, a, false, b1,
                                                   (short)0, c1, false, false);
    }
    __syncthreads();
  }

  // ---- Epilogue: bias + tanh + recurrent elementwise updates ----
  const int mbase = m0 + (lane >> 4) * 8;        // C/D: lanes 16-31 hold M+8
  const bool harmonic = (n0 < PP);               // uniform per block

#pragma unroll
  for (int tt = 0; tt < 2; ++tt) {
    v8f cacc = tt ? c1 : c0;
    const int n  = n0 + tt * 16 + (lane & 15);
    const float bv = bias[n];
#pragma unroll
    for (int r = 0; r < 8; ++r) {
      const int gm = mbase + r;
      const float f = tanhf(cacc[r] + bv);
      const size_t sidx  = (size_t)gm * NH + n;
      const size_t obase = ((size_t)gm * TT + t) * PP;
      if (harmonic) {
        const int p = n;
        const float hz_old = hzf[(size_t)gm * PP + p];
        const float hy_old = hyf[sidx];
        const float hz_new = hz_old + DT_C * f - GAMMA_C * hy_old - EPS_C * hz_old;
        const float hy_new = hy_old + DT_C * hz_new;
        hzf[(size_t)gm * PP + p] = hz_new;
        hyf[sidx]     = hy_new;
        hyb_out[sidx] = (__bf16)hy_new;
        out[obase + p]           = hy_new;       // hy_m
        out[OUT_SEC + obase + p] = hz_new;       // hz
      } else {
        const int p = n - PP;
        const float u     = hyf[sidx];
        const float spike = (u > 1.0f) ? 1.0f : 0.0f;
        const float ur    = (u > 1.0f) ? 0.0f : u;
        const float un    = ur + DT_C * (-ur + f);   // R*C = 1
        hyf[sidx]     = un;
        hyb_out[sidx] = (__bf16)un;
        out[2 * OUT_SEC + obase + p] = un;       // hy_u
        out[3 * OUT_SEC + obase + p] = spike;    // spikes
      }
    }
  }
}

extern "C" void kernel_launch(void* const* d_in, const int* in_sizes, int n_in,
                              void* d_out, int out_size, void* d_ws, size_t ws_size,
                              hipStream_t stream) {
  const float* x    = (const float*)d_in[0];   // [64, 512, 256]
  const float* x2h  = (const float*)d_in[1];   // [256, 2048]
  const float* h2h  = (const float*)d_in[2];   // [2048, 2048]
  const float* bias = (const float*)d_in[3];   // [2048]
  float* out = (float*)d_out;

  char* ws = (char*)d_ws;
  __bf16* wpack = (__bf16*)(ws);
  __bf16* hyb0  = (__bf16*)(ws + WPACK_BYTES);
  __bf16* hyb1  = (__bf16*)(ws + WPACK_BYTES + HYB_BYTES);
  float*  hyf   = (float*) (ws + WPACK_BYTES + 2 * HYB_BYTES);
  float*  hzf   = (float*) (ws + WPACK_BYTES + 2 * HYB_BYTES + HYF_BYTES);

  // One-time (per call) weight packing + state init.
  {
    size_t total = (size_t)NBLOCKS * KTILES * 4 * 512;   // 4,718,592
    int blocks = (int)((total + 255) / 256);             // 18,432
    pack_weights_kernel<<<blocks, 256, 0, stream>>>(x2h, h2h, wpack);
    init_state_kernel<<<(BB * NH + 255) / 256, 256, 0, stream>>>(hyf, hzf, hyb0);
  }

  // Sequential recurrence: one kernel per timestep, hy bf16 ping-pong.
  for (int t = 0; t < TT; ++t) {
    const __bf16* hin  = (t & 1) ? hyb1 : hyb0;
    __bf16*       hout = (t & 1) ? hyb0 : hyb1;
    step_kernel<<<NBLOCKS, 256, 0, stream>>>(t, x, wpack, bias, hin, hout,
                                             hyf, hzf, out);
  }
}
